// GPTTransformer_26946624815249
// MI455X (gfx1250) — matrix operand order repaired and checked by reading, not compiled, not run on hardware
//
#include <hip/hip_runtime.h>
#include <hip/hip_bf16.h>

// ---------------------------------------------------------------------------
// GPT forward for MI455X (gfx1250): bf16 WMMA GEMMs + flash attention.
// Round 4: async global->LDS staging via inline asm
// (global_load_async_to_lds_b128 + s_wait_asynccnt), double-buffered tiles.
// ---------------------------------------------------------------------------

typedef __attribute__((ext_vector_type(16))) __bf16 v16bf;
typedef __attribute__((ext_vector_type(8)))  float  v8f;

#define GPT_V  32000
#define GPT_T  1024
#define GPT_C  768
#define GPT_H  12
#define GPT_L  6
#define GPT_HS 64
#define GPT_B  4
#define GPT_M  (GPT_B * GPT_T)   // 4096 rows of the activation matrix

// Feature-detect: if the toolchain declares the async builtin, its assembler
// understands the gfx1250 async-to-LDS instructions (we use inline asm to
// avoid depending on the builtin's exact signature).
#if defined(__has_builtin)
#if __has_builtin(__builtin_amdgcn_global_load_async_to_lds_b128)
#define HAVE_ASYNC_LDS 1
#endif
#endif

__device__ __forceinline__ __bf16 f2bf(float f) {
    unsigned u = __builtin_bit_cast(unsigned, f);
    unsigned r = (u + 0x7FFFu + ((u >> 16) & 1u)) >> 16;
    unsigned short s = (unsigned short)r;
    return __builtin_bit_cast(__bf16, s);
}

__device__ __forceinline__ v8f wmma_bf16(v16bf a, v16bf b, v8f c) {
    return __builtin_amdgcn_wmma_f32_16x16x32_bf16(
        /*neg_a=*/false, a, /*neg_b=*/false, b,
        /*c_mod=*/(short)0, c, /*reuse_a=*/false, /*reuse_b=*/false);
}

#ifdef HAVE_ASYNC_LDS
// per-lane 16B global -> LDS async copy (tracked with ASYNCcnt)
__device__ __forceinline__ void async_cp16(const void* g, void* l) {
    unsigned lds_off = (unsigned)(unsigned long long)l;   // low 32 bits = LDS offset
    asm volatile("global_load_async_to_lds_b128 %0, %1, off"
                 :: "v"(lds_off), "v"((unsigned long long)g)
                 : "memory");
}
__device__ __forceinline__ void wait_async0() {
    asm volatile("s_wait_asynccnt 0x0" ::: "memory");
}
#endif

// ---------------------------------------------------------------------------
// f32 -> bf16 conversion (weights, once per launch)
// ---------------------------------------------------------------------------
__global__ void cvt_f32_bf16(const float* __restrict__ in, __bf16* __restrict__ out, size_t n) {
    size_t i = (size_t)blockIdx.x * blockDim.x + threadIdx.x;
    if (i < n) out[i] = f2bf(in[i]);
}

// ---------------------------------------------------------------------------
// Embedding: x[b,t,c] = tok_emb[tokens[b,t]][c] + pos_emb[t][c]   (fp32)
// ---------------------------------------------------------------------------
__global__ void embed_kernel(const int* __restrict__ tokens,
                             const float* __restrict__ tok_emb,
                             const float* __restrict__ pos_emb,
                             float* __restrict__ x) {
    size_t i = (size_t)blockIdx.x * blockDim.x + threadIdx.x;
    size_t total = (size_t)GPT_M * GPT_C;
    if (i >= total) return;
    int bt = (int)(i / GPT_C);
    int c  = (int)(i % GPT_C);
    int t  = bt % GPT_T;
    int tok = tokens[bt];
    x[i] = tok_emb[(size_t)tok * GPT_C + c] + pos_emb[(size_t)t * GPT_C + c];
}

// ---------------------------------------------------------------------------
// LayerNorm: one row (C=768) per block, fp32 in, bf16 out (feeds WMMA GEMM)
// ---------------------------------------------------------------------------
__global__ __launch_bounds__(256)
void layernorm_kernel(const float* __restrict__ x,
                      const float* __restrict__ g,
                      const float* __restrict__ b,
                      __bf16* __restrict__ out) {
    __shared__ float red[256];
    const int row = blockIdx.x;
    const float* xr = x + (size_t)row * GPT_C;
    const int tid = threadIdx.x;

    float s = 0.f;
    for (int c = tid; c < GPT_C; c += 256) s += xr[c];
    red[tid] = s; __syncthreads();
    for (int o = 128; o > 0; o >>= 1) {
        if (tid < o) red[tid] += red[tid + o];
        __syncthreads();
    }
    float mu = red[0] * (1.0f / GPT_C);
    __syncthreads();

    float vs = 0.f;
    for (int c = tid; c < GPT_C; c += 256) { float d = xr[c] - mu; vs += d * d; }
    red[tid] = vs; __syncthreads();
    for (int o = 128; o > 0; o >>= 1) {
        if (tid < o) red[tid] += red[tid + o];
        __syncthreads();
    }
    float inv = rsqrtf(red[0] * (1.0f / GPT_C) + 1e-5f);

    __bf16* orow = out + (size_t)row * GPT_C;
    for (int c = tid; c < GPT_C; c += 256)
        orow[c] = f2bf((xr[c] - mu) * inv * g[c] + b[c]);
}

// ---------------------------------------------------------------------------
// WMMA GEMM: C[MxN] = A[MxK](bf16) * B[KxN](bf16) (+bias)(+relu)(+residual)
// Block tile 128x128, BK=32, 8 waves (2x4), each wave 64x32 = 4x2 WMMA tiles.
// Double-buffered LDS: A tile row-major (async-copied), B tile stored
// TRANSPOSED (BsT[n][k]) so every fragment loads as ds_load_b128 pairs.
// ---------------------------------------------------------------------------
#define BM 128
#define BN 128
#define BKK 32

template <bool BIAS, bool RELU, bool RESID, bool OUTB>
__global__ __launch_bounds__(256)
void gemm_bf16_kernel(const __bf16* __restrict__ A,
                      const __bf16* __restrict__ B,
                      const float* __restrict__ bias,
                      float* __restrict__ outf,
                      __bf16* __restrict__ outb,
                      int M, int N, int K) {
    __shared__ __bf16 As [2][BM][BKK + 8];   // row-major, 80B row stride
    __shared__ __bf16 BsT[2][BN][BKK + 8];   // transposed: [n][k]

    const int tid  = threadIdx.x;
    const int lane = tid & 31;
    const int wave = tid >> 5;       // 0..7
    const int wm   = wave >> 2;      // 0..1 (M dir, 64 rows each)
    const int wn   = wave & 3;       // 0..3 (N dir, 32 cols each)
    const int half = lane >> 4;
    const int l16  = lane & 15;

    const int mBlock = blockIdx.y * BM;
    const int nBlock = blockIdx.x * BN;

    v8f acc[4][2];
#pragma unroll
    for (int i = 0; i < 4; ++i)
#pragma unroll
        for (int j = 0; j < 2; ++j)
#pragma unroll
            for (int e = 0; e < 8; ++e) acc[i][j][e] = 0.f;

    // ---- tile staging ------------------------------------------------------
    auto stageA = [&](int buf, int k0) {
#ifdef HAVE_ASYNC_LDS
        // 128x32 bf16 = 512 x 16B chunks; 256 threads -> 2 async b128 each
#pragma unroll
        for (int rnd = 0; rnd < 2; ++rnd) {
            int chunk = tid + rnd * 256;
            int r = chunk >> 2, c = (chunk & 3) * 8;
            async_cp16(A + (size_t)(mBlock + r) * K + k0 + c, &As[buf][r][c]);
        }
#else
#pragma unroll
        for (int i = tid; i < (BM * BKK) / 2; i += 256) {
            int r = (i * 2) / BKK, c = (i * 2) % BKK;
            *(unsigned*)&As[buf][r][c] =
                *(const unsigned*)(A + (size_t)(mBlock + r) * K + k0 + c);
        }
#endif
    };
    auto stageB = [&](int buf, int k0) {
        // coalesced dword loads (2 consecutive n), transposed u16 stores
#pragma unroll
        for (int i = tid; i < (BKK * BN) / 2; i += 256) {
            int r = (i * 2) / BN, c = (i * 2) % BN;   // r = k, c = n
            unsigned u = *(const unsigned*)(B + (size_t)(k0 + r) * N + nBlock + c);
            BsT[buf][c][r]     = __builtin_bit_cast(__bf16, (unsigned short)(u & 0xFFFFu));
            BsT[buf][c + 1][r] = __builtin_bit_cast(__bf16, (unsigned short)(u >> 16));
        }
    };

    const int nk = K / BKK;
    stageA(0, 0);
    stageB(0, 0);
#ifdef HAVE_ASYNC_LDS
    wait_async0();
#endif
    __syncthreads();

    int p = 0;
    for (int kt = 0; kt < nk; ++kt) {
        if (kt + 1 < nk) {            // stage next tile into the other buffer
            stageA(p ^ 1, (kt + 1) * BKK);
            stageB(p ^ 1, (kt + 1) * BKK);
        }

        // B fragments: lane holds column n; element e -> K = 16*half + e (contiguous)
        v16bf bf[2];
#pragma unroll
        for (int j = 0; j < 2; ++j) {
            int col = wn * 32 + j * 16 + l16;
#pragma unroll
            for (int e = 0; e < 16; ++e) bf[j][e] = BsT[p][col][16 * half + e];
        }
        // A fragments: lane holds row m; element e -> K = 16*half + e (contiguous)
        v16bf af[4];
#pragma unroll
        for (int i = 0; i < 4; ++i) {
            int row = wm * 64 + i * 16 + l16;
#pragma unroll
            for (int e = 0; e < 16; ++e) af[i][e] = As[p][row][16 * half + e];
        }
#pragma unroll
        for (int i = 0; i < 4; ++i)
#pragma unroll
            for (int j = 0; j < 2; ++j)
                acc[i][j] = wmma_bf16(af[i], bf[j], acc[i][j]);

#ifdef HAVE_ASYNC_LDS
        wait_async0();
#endif
        __syncthreads();
        p ^= 1;
    }

    // epilogue: C layout lane=(col n=l16), VGPR r -> row = r + 8*half
#pragma unroll
    for (int i = 0; i < 4; ++i) {
#pragma unroll
        for (int j = 0; j < 2; ++j) {
            int col = nBlock + wn * 32 + j * 16 + l16;
            float bv = BIAS ? bias[col] : 0.f;
#pragma unroll
            for (int r = 0; r < 8; ++r) {
                int row = mBlock + wm * 64 + i * 16 + r + 8 * half;
                float v = acc[i][j][r] + bv;
                if (RELU) v = fmaxf(v, 0.f);
                size_t idx = (size_t)row * N + col;
                if (OUTB) {
                    outb[idx] = f2bf(v);
                } else {
                    if (RESID) v += outf[idx];
                    outf[idx] = v;
                }
            }
        }
    }
}

// ---------------------------------------------------------------------------
// Flash attention (causal): grid (T/64, H, B), 128 threads = 4 waves.
// Each wave owns 16 query rows; streams 32-key tiles of K/V through
// double-buffered LDS (K async-copied row-major; V transposed manually).
// QK^T: 4 WMMAs; PV: 4 WMMAs. P staging is wave-private -> wave_barrier only.
// ---------------------------------------------------------------------------
__global__ __launch_bounds__(128)
void attn_kernel(const __bf16* __restrict__ q,
                 const __bf16* __restrict__ k,
                 const __bf16* __restrict__ v,
                 __bf16* __restrict__ o) {
    __shared__ __bf16 Ks [2][32][GPT_HS];        // [key][d]
    __shared__ __bf16 VsT[2][GPT_HS][32 + 8];    // [d][key], 80B row stride
    __shared__ __bf16 Pb[4][16][40];             // per-wave P staging, 80B rows

    const int b    = blockIdx.z;
    const int h    = blockIdx.y;
    const int qtil = blockIdx.x * 64;
    const int tid  = threadIdx.x;
    const int lane = tid & 31;
    const int wave = tid >> 5;
    const int half = lane >> 4;
    const int l16  = lane & 15;
    const int qrow0 = qtil + wave * 16;
    const float scale = 0.125f;  // HS^-0.5

    // Q fragments (16x64 = 2 A-frags of 16x32)
    v16bf qf[2];
    {
        int row = qrow0 + l16;
        size_t base = ((size_t)(b * GPT_T + row) * GPT_H + h) * GPT_HS;
#pragma unroll
        for (int f = 0; f < 2; ++f)
#pragma unroll
            for (int e = 0; e < 16; ++e)
                qf[f][e] = q[base + f * 32 + 16 * half + e];
    }

    v8f oacc[4];
    float mI[8], lI[8];
#pragma unroll
    for (int j = 0; j < 4; ++j)
#pragma unroll
        for (int e = 0; e < 8; ++e) oacc[j][e] = 0.f;
#pragma unroll
    for (int r = 0; r < 8; ++r) { mI[r] = -1e30f; lI[r] = 0.f; }

    auto stageKV = [&](int buf, int kbase) {
#ifdef HAVE_ASYNC_LDS
        // K tile 32x64 bf16 = 256 x 16B chunks; 128 threads -> 2 async b128
#pragma unroll
        for (int rnd = 0; rnd < 2; ++rnd) {
            int chunk = tid + rnd * 128;
            int r = chunk >> 3, c = (chunk & 7) * 8;
            async_cp16(k + ((size_t)(b * GPT_T + kbase + r) * GPT_H + h) * GPT_HS + c,
                       &Ks[buf][r][c]);
        }
#else
#pragma unroll
        for (int i = tid; i < (32 * GPT_HS) / 2; i += 128) {
            int r = (i * 2) / GPT_HS, c = (i * 2) % GPT_HS;
            *(unsigned*)&Ks[buf][r][c] = *(const unsigned*)(
                k + ((size_t)(b * GPT_T + kbase + r) * GPT_H + h) * GPT_HS + c);
        }
#endif
        // V tile: coalesced dword loads, transposed u16 stores
#pragma unroll
        for (int i = tid; i < (32 * GPT_HS) / 2; i += 128) {
            int r = (i * 2) / GPT_HS, c = (i * 2) % GPT_HS;   // r = key, c = d
            unsigned uv = *(const unsigned*)(
                v + ((size_t)(b * GPT_T + kbase + r) * GPT_H + h) * GPT_HS + c);
            VsT[buf][c][r]     = __builtin_bit_cast(__bf16, (unsigned short)(uv & 0xFFFFu));
            VsT[buf][c + 1][r] = __builtin_bit_cast(__bf16, (unsigned short)(uv >> 16));
        }
    };

    const int ktiles = (qtil + 64) / 32;
    stageKV(0, 0);
#ifdef HAVE_ASYNC_LDS
    wait_async0();
#endif
    __syncthreads();

    int p = 0;
    for (int kt = 0; kt < ktiles; ++kt) {
        const int kbase = kt * 32;
        if (kt + 1 < ktiles) stageKV(p ^ 1, kbase + 32);

        // S(16x32) = Q(16x64) @ K^T(64x32)
        v8f s[2];
#pragma unroll
        for (int jj = 0; jj < 2; ++jj) {
#pragma unroll
            for (int e = 0; e < 8; ++e) s[jj][e] = 0.f;
            int key = jj * 16 + l16;
#pragma unroll
            for (int f = 0; f < 2; ++f) {
                v16bf bf;  // B[d][key] = Ks[key][d]; element e -> d = f*32 + 16*half + e
#pragma unroll
                for (int e = 0; e < 16; ++e) bf[e] = Ks[p][key][f * 32 + 16 * half + e];
                s[jj] = wmma_bf16(qf[f], bf, s[jj]);
            }
        }

        // scale + causal mask + online softmax (rows r+8*half, cols across 16 lanes)
#pragma unroll
        for (int r = 0; r < 8; ++r) {
            int qi = qrow0 + r + 8 * half;
            float mx = -1e30f;
#pragma unroll
            for (int jj = 0; jj < 2; ++jj) {
                int key = kbase + jj * 16 + l16;
                float xv = s[jj][r] * scale;
                if (key > qi) xv = -1e30f;
                s[jj][r] = xv;
                mx = fmaxf(mx, xv);
            }
            for (int off = 1; off < 16; off <<= 1)
                mx = fmaxf(mx, __shfl_xor(mx, off, 32));
            float mNew  = fmaxf(mI[r], mx);
            float alpha = __expf(mI[r] - mNew);
            float rs = 0.f;
#pragma unroll
            for (int jj = 0; jj < 2; ++jj) {
                float p2 = __expf(s[jj][r] - mNew);
                Pb[wave][r + 8 * half][jj * 16 + l16] = f2bf(p2);
                rs += p2;
            }
            for (int off = 1; off < 16; off <<= 1)
                rs += __shfl_xor(rs, off, 32);
            lI[r] = lI[r] * alpha + rs;
            mI[r] = mNew;
#pragma unroll
            for (int j = 0; j < 4; ++j) oacc[j][r] *= alpha;
        }
        // P staging is wave-private and same-wave LDS ops are in-order:
        // a scheduling fence is enough (no block barrier).
        __builtin_amdgcn_wave_barrier();

        // O(16x64) += P(16x32) @ V(32x64)
        v16bf pf;
#pragma unroll
        for (int e = 0; e < 16; ++e)
            pf[e] = Pb[wave][l16][16 * half + e];
#pragma unroll
        for (int j = 0; j < 4; ++j) {
            v16bf vf;  // B[key][d] = VsT[d][key]; element e -> key = 16*half + e
            int d = j * 16 + l16;
#pragma unroll
            for (int e = 0; e < 16; ++e) vf[e] = VsT[p][d][16 * half + e];
            oacc[j] = wmma_bf16(pf, vf, oacc[j]);
        }

#ifdef HAVE_ASYNC_LDS
        wait_async0();
#endif
        __syncthreads();   // next tile may overwrite the other buffer
        p ^= 1;
    }

    // write O / l  (bf16, [B,T,H,HS] layout == [M, C] for the next GEMM)
#pragma unroll
    for (int j = 0; j < 4; ++j)
#pragma unroll
        for (int r = 0; r < 8; ++r) {
            int row = qrow0 + r + 8 * half;
            int d   = j * 16 + l16;
            float val = oacc[j][r] / lI[r];
            o[((size_t)(b * GPT_T + row) * GPT_H + h) * GPT_HS + d] = f2bf(val);
        }
}

// ---------------------------------------------------------------------------
// Host orchestration
// ---------------------------------------------------------------------------
static void launch_cvt(const float* src, __bf16* dst, size_t n, hipStream_t s) {
    int blk = 256;
    int grid = (int)((n + blk - 1) / blk);
    cvt_f32_bf16<<<grid, blk, 0, s>>>(src, dst, n);
}

extern "C" void kernel_launch(void* const* d_in, const int* in_sizes, int n_in,
                              void* d_out, int out_size, void* d_ws, size_t ws_size,
                              hipStream_t stream) {
    (void)in_sizes; (void)n_in; (void)out_size; (void)ws_size;
    const int*   tokens  = (const int*)d_in[0];
    const float* tok_emb = (const float*)d_in[1];
    const float* pos_emb = (const float*)d_in[2];
    const float* ln1_g   = (const float*)d_in[3];
    const float* ln1_b   = (const float*)d_in[4];
    const float* wq      = (const float*)d_in[5];
    const float* wk      = (const float*)d_in[6];
    const float* wv      = (const float*)d_in[7];
    const float* wo      = (const float*)d_in[8];
    const float* bo      = (const float*)d_in[9];
    const float* ln2_g   = (const float*)d_in[10];
    const float* ln2_b   = (const float*)d_in[11];
    const float* w1      = (const float*)d_in[12];
    const float* b1      = (const float*)d_in[13];
    const float* w2      = (const float*)d_in[14];
    const float* b2      = (const float*)d_in[15];
    const float* lnf_g   = (const float*)d_in[16];
    const float* lnf_b   = (const float*)d_in[17];
    const float* lm_w    = (const float*)d_in[18];
    const float* lm_b    = (const float*)d_in[19];
    float* logits = (float*)d_out;

    const size_t C = GPT_C, C4 = 4 * GPT_C, M = GPT_M, V = GPT_V, Lc = GPT_L;

    char* ws = (char*)d_ws;
    size_t off = 0;
    auto carve = [&](size_t bytes) -> void* {
        void* p = ws + off;
        off = (off + bytes + 255) & ~(size_t)255;
        return p;
    };
    float*  x     = (float*)  carve(M * C * sizeof(float));
    __bf16* hbf   = (__bf16*) carve(M * C * 2);
    __bf16* qbf   = (__bf16*) carve(M * C * 2);
    __bf16* kbf   = (__bf16*) carve(M * C * 2);
    __bf16* vbf   = (__bf16*) carve(M * C * 2);
    __bf16* obf   = (__bf16*) carve(M * C * 2);
    __bf16* midbf = (__bf16*) carve(M * C4 * 2);
    __bf16* xfbf  = (__bf16*) carve(M * C * 2);
    __bf16* wq_bf = (__bf16*) carve(Lc * C * C * 2);
    __bf16* wk_bf = (__bf16*) carve(Lc * C * C * 2);
    __bf16* wv_bf = (__bf16*) carve(Lc * C * C * 2);
    __bf16* wo_bf = (__bf16*) carve(Lc * C * C * 2);
    __bf16* w1_bf = (__bf16*) carve(Lc * C * C4 * 2);
    __bf16* w2_bf = (__bf16*) carve(Lc * C4 * C * 2);
    __bf16* lmw_bf= (__bf16*) carve(C * V * 2);

    // weights -> bf16 (every call; deterministic)
    launch_cvt(wq, wq_bf, Lc * C * C, stream);
    launch_cvt(wk, wk_bf, Lc * C * C, stream);
    launch_cvt(wv, wv_bf, Lc * C * C, stream);
    launch_cvt(wo, wo_bf, Lc * C * C, stream);
    launch_cvt(w1, w1_bf, Lc * C * C4, stream);
    launch_cvt(w2, w2_bf, Lc * C4 * C, stream);
    launch_cvt(lm_w, lmw_bf, C * V, stream);

    // embedding
    {
        size_t total = M * C;
        embed_kernel<<<(int)((total + 255) / 256), 256, 0, stream>>>(tokens, tok_emb, pos_emb, x);
    }

    dim3 gC((int)(C / BN),  (int)(M / BM));   // N=768
    dim3 g4((int)(C4 / BN), (int)(M / BM));   // N=3072
    dim3 gV((int)(V / BN),  (int)(M / BM));   // N=32000
    dim3 gA((int)(GPT_T / 64), GPT_H, GPT_B);

    for (int l = 0; l < GPT_L; ++l) {
        layernorm_kernel<<<(int)M, 256, 0, stream>>>(x, ln1_g + l * C, ln1_b + l * C, hbf);

        // q/k/v = h @ w   (bf16 out, no bias)
        gemm_bf16_kernel<false,false,false,true><<<gC, 256, 0, stream>>>(
            hbf, wq_bf + (size_t)l * C * C, nullptr, nullptr, qbf, (int)M, (int)C, (int)C);
        gemm_bf16_kernel<false,false,false,true><<<gC, 256, 0, stream>>>(
            hbf, wk_bf + (size_t)l * C * C, nullptr, nullptr, kbf, (int)M, (int)C, (int)C);
        gemm_bf16_kernel<false,false,false,true><<<gC, 256, 0, stream>>>(
            hbf, wv_bf + (size_t)l * C * C, nullptr, nullptr, vbf, (int)M, (int)C, (int)C);

        attn_kernel<<<gA, 128, 0, stream>>>(qbf, kbf, vbf, obf);

        // x = o @ wo + bo + x   (fp32 residual)
        gemm_bf16_kernel<true,false,true,false><<<gC, 256, 0, stream>>>(
            obf, wo_bf + (size_t)l * C * C, bo + l * C, x, nullptr, (int)M, (int)C, (int)C);

        layernorm_kernel<<<(int)M, 256, 0, stream>>>(x, ln2_g + l * C, ln2_b + l * C, hbf);

        // mid = relu(h2 @ w1 + b1)   (bf16 out)
        gemm_bf16_kernel<true,true,false,true><<<g4, 256, 0, stream>>>(
            hbf, w1_bf + (size_t)l * C * C4, b1 + l * C4, nullptr, midbf, (int)M, (int)C4, (int)C);
        // x = mid @ w2 + b2 + x   (fp32 residual)
        gemm_bf16_kernel<true,false,true,false><<<gC, 256, 0, stream>>>(
            midbf, w2_bf + (size_t)l * C4 * C, b2 + l * C, x, nullptr, (int)M, (int)C, (int)C4);
    }

    layernorm_kernel<<<(int)M, 256, 0, stream>>>(x, lnf_g, lnf_b, xfbf);

    // logits = xf @ lm_w + lm_b   (fp32 out)
    gemm_bf16_kernel<true,false,false,false><<<gV, 256, 0, stream>>>(
        xfbf, lmw_bf, lm_b, logits, nullptr, (int)M, (int)V, (int)C);
}
